// MambaTower_66374424592997
// MI455X (gfx1250) — compile-verified, matrix operand order checked by reading, not run on hardware
//
#include <hip/hip_runtime.h>

// ---- problem constants (match reference) ----
#define B_   2
#define L_   2048
#define E_   1024
#define NL_  2
#define DI_  2048
#define DS_  16
#define DC_  4
#define DTR_ 64
#define ML_  (B_*L_)      // 4096 flattened rows

typedef __attribute__((ext_vector_type(16))) __bf16 v16bf;
typedef __attribute__((ext_vector_type(8)))  float  v8f;

struct alignas(16) U4 { unsigned int x, y, z, w; };
union Frag { v16bf v; U4 q[2]; };

__device__ __forceinline__ unsigned short f2bf(float f) {
  unsigned int u = __builtin_bit_cast(unsigned int, f);
  u += 0x7FFFu + ((u >> 16) & 1u);          // round-to-nearest-even
  return (unsigned short)(u >> 16);
}
__device__ __forceinline__ float sigm_(float x) { return 1.f / (1.f + __expf(-x)); }

// =====================================================================
// WMMA GEMM:  Cf[M,N] (f32) = A[M,K](bf16,row-major,lda) * Bw[N,K]^T (bf16,ldb)
// wave tile 32x64 (2 A frags x 4 B frags, 8 v_wmma per K-step of 32)
// block = 8 waves -> 64x256 tile.  epilogue: 0=store, 1=softplus(x+bias[n])
// =====================================================================
__global__ __launch_bounds__(256) void gemm_bf16_nt(
    const unsigned short* __restrict__ A, int lda,
    const unsigned short* __restrict__ Bw, int ldb,
    float* __restrict__ Cf, unsigned short* __restrict__ Cbf, int ldc,
    int M, int N, int K, const float* __restrict__ bias, int epilogue)
{
  const int lane = threadIdx.x & 31;
  const int wave = threadIdx.x >> 5;
  const int wm = wave >> 2, wn = wave & 3;
  const int bm = blockIdx.x * 64 + wm * 32;
  const int bn = blockIdx.y * 256 + wn * 64;
  const int half = lane >> 4, l16 = lane & 15;

  v8f acc[2][4];
  const v8f zero = {0.f,0.f,0.f,0.f,0.f,0.f,0.f,0.f};
#pragma unroll
  for (int i = 0; i < 2; ++i)
#pragma unroll
    for (int j = 0; j < 4; ++j) acc[i][j] = zero;

  // per-lane fragment base pointers (16-bit A/B layout: lane<16 -> K 0..7,16..23)
  const unsigned short* Ab0 = A + (size_t)(bm + l16) * lda + half * 8;
  const unsigned short* Ab1 = Ab0 + (size_t)16 * lda;
  const unsigned short* Bb[4];
#pragma unroll
  for (int j = 0; j < 4; ++j) {
    int c = bn + j * 16 + l16;
    if (c >= N) c = N - 1;                    // clamp (stores are guarded)
    Bb[j] = Bw + (size_t)c * ldb + half * 8;
  }

  for (int k0 = 0; k0 < K; k0 += 32) {
    Frag a0, a1, bf[4];
    a0.q[0] = *(const U4*)(Ab0 + k0);  a0.q[1] = *(const U4*)(Ab0 + k0 + 16);
    a1.q[0] = *(const U4*)(Ab1 + k0);  a1.q[1] = *(const U4*)(Ab1 + k0 + 16);
#pragma unroll
    for (int j = 0; j < 4; ++j) {
      bf[j].q[0] = *(const U4*)(Bb[j] + k0);
      bf[j].q[1] = *(const U4*)(Bb[j] + k0 + 16);
    }
    if (k0 + 32 < K) {                        // gfx1250 global_prefetch_b8
      __builtin_prefetch(Ab0 + k0 + 32, 0, 1);
      __builtin_prefetch(Bb[0] + k0 + 32, 0, 1);
    }
#pragma unroll
    for (int j = 0; j < 4; ++j) {
      acc[0][j] = __builtin_amdgcn_wmma_f32_16x16x32_bf16(
          false, a0.v, false, bf[j].v, (short)0, acc[0][j], false, false);
      acc[1][j] = __builtin_amdgcn_wmma_f32_16x16x32_bf16(
          false, a1.v, false, bf[j].v, (short)0, acc[1][j], false, false);
    }
  }

  // C/D layout: lanes 0-15: VGPR r -> (M=r, N=lane); lanes 16-31: (M=r+8, N=lane-16)
#pragma unroll
  for (int i = 0; i < 2; ++i)
#pragma unroll
    for (int j = 0; j < 4; ++j) {
      int col = bn + j * 16 + l16;
      if (col >= N) continue;
#pragma unroll
      for (int r = 0; r < 8; ++r) {
        int row = bm + i * 16 + half * 8 + r;
        float v = acc[i][j][r];
        if (epilogue == 1) {                  // softplus(x + dt_bias)
          v += bias[col];
          v = (v > 20.f) ? v : log1pf(__expf(v));
        }
        if (Cf)  Cf [(size_t)row * ldc + col] = v;
        if (Cbf) Cbf[(size_t)row * ldc + col] = f2bf(v);
      }
    }
}

// ---- f32 -> bf16 weight conversion ----
__global__ __launch_bounds__(256) void cvt_bf16_k(const float* __restrict__ in,
                                                  unsigned short* __restrict__ out, int n) {
  int i = blockIdx.x * 256 + threadIdx.x;
  if (i < n) out[i] = f2bf(in[i]);
}

// ---- h = x + pos_embed ; also bf16 mirror ----
__global__ __launch_bounds__(256) void addpos_k(const float* __restrict__ x,
    const float* __restrict__ pos, float* __restrict__ hf,
    unsigned short* __restrict__ hb, int n) {
  int idx = blockIdx.x * 256 + threadIdx.x;
  if (idx >= n) return;
  int e = idx % E_;
  int l = (idx / E_) % L_;
  float v = x[idx] + pos[l * E_ + e];
  hf[idx] = v; hb[idx] = f2bf(v);
}

// ---- causal depthwise conv (DC=4) + SiLU; reads xc half of xz ----
__global__ __launch_bounds__(256) void conv_silu_k(const float* __restrict__ xz,
    const float* __restrict__ cw, const float* __restrict__ cb,
    float* __restrict__ xcf, unsigned short* __restrict__ xcb) {
  int idx = blockIdx.x * 256 + threadIdx.x;
  if (idx >= ML_ * DI_) return;
  int d  = idx % DI_;
  int bl = idx / DI_;
  int l  = bl % L_;
  int b  = bl / L_;
  float acc = 0.f;
#pragma unroll
  for (int k = 0; k < DC_; ++k) {
    int ls = l + k - (DC_ - 1);
    if (ls >= 0) acc += cw[d * DC_ + k] * xz[((size_t)(b * L_ + ls)) * (2 * DI_) + d];
  }
  acc += cb[d];
  float s = acc * sigm_(acc);
  xcf[idx] = s; xcb[idx] = f2bf(s);
}

// ---- selective scan: one lane per channel d, DS=16 state in VGPRs ----
// fuses D-skip and the SiLU(z) gate; emits bf16 y for the out_proj WMMA.
__global__ __launch_bounds__(256) void scan_k(const float* __restrict__ delta,
    const float* __restrict__ xdbl, const float* __restrict__ xcf,
    const float* __restrict__ xz, const float* __restrict__ A_log,
    const float* __restrict__ Dp, unsigned short* __restrict__ yb) {
  __shared__ float sBC[32];                   // Bm[0..15], C[0..15] for current l
  const int b = blockIdx.y;
  const int d = blockIdx.x * 256 + threadIdx.x;
  float Av[DS_], hs[DS_];
#pragma unroll
  for (int n = 0; n < DS_; ++n) { Av[n] = -__expf(A_log[d * DS_ + n]); hs[n] = 0.f; }
  const float dsk = Dp[d];
  for (int l = 0; l < L_; ++l) {
    const size_t row = (size_t)(b * L_ + l);
    if (threadIdx.x < 32) sBC[threadIdx.x] = xdbl[row * 96 + DTR_ + threadIdx.x];
    __syncthreads();
    const float dl = delta[row * DI_ + d];
    const float u  = xcf[row * DI_ + d];
    const float du = dl * u;
    float y = 0.f;
#pragma unroll
    for (int n = 0; n < DS_; ++n) {
      hs[n] = __expf(dl * Av[n]) * hs[n] + du * sBC[n];
      y += hs[n] * sBC[DS_ + n];
    }
    const float z = xz[row * (2 * DI_) + DI_ + d];
    const float o = (y + u * dsk) * (z * sigm_(z));
    yb[row * DI_ + d] = f2bf(o);
    __syncthreads();
  }
}

// ---- fused residual + RMSNorm; updates h (f32+bf16), optional final output ----
__global__ __launch_bounds__(256) void rmsnorm_k(const float* __restrict__ outp,
    float* __restrict__ hf, unsigned short* __restrict__ hb,
    const float* __restrict__ nw, float* __restrict__ final_out) {
  __shared__ float red[8];
  __shared__ float rstd_s;
  const size_t row = blockIdx.x;
  const int lane = threadIdx.x & 31, wv = threadIdx.x >> 5;
  float ss = 0.f;
  for (int e = threadIdx.x; e < E_; e += 256) {
    float v = outp[row * E_ + e] + hf[row * E_ + e];
    ss += v * v;
  }
#pragma unroll
  for (int o = 16; o > 0; o >>= 1) ss += __shfl_down(ss, o, 32);
  if (lane == 0) red[wv] = ss;
  __syncthreads();
  if (threadIdx.x == 0) {
    float t = 0.f;
#pragma unroll
    for (int i = 0; i < 8; ++i) t += red[i];
    rstd_s = rsqrtf(t / (float)E_ + 1e-6f);
  }
  __syncthreads();
  const float rstd = rstd_s;
  for (int e = threadIdx.x; e < E_; e += 256) {
    float v  = outp[row * E_ + e] + hf[row * E_ + e];
    float hn = v * rstd * nw[e];
    hf[row * E_ + e] = hn;
    hb[row * E_ + e] = f2bf(hn);
    if (final_out) final_out[row * E_ + e] = hn;
  }
}

// =====================================================================
extern "C" void kernel_launch(void* const* d_in, const int* in_sizes, int n_in,
                              void* d_out, int out_size, void* d_ws, size_t ws_size,
                              hipStream_t stream) {
  (void)in_sizes; (void)n_in; (void)out_size; (void)ws_size;
  const float* x    = (const float*)d_in[0];
  const float* pos  = (const float*)d_in[1];
  const float* w_in = (const float*)d_in[2];
  const float* cw   = (const float*)d_in[3];
  const float* cb   = (const float*)d_in[4];
  const float* w_x  = (const float*)d_in[5];
  const float* w_dt = (const float*)d_in[6];
  const float* dtb  = (const float*)d_in[7];
  const float* alog = (const float*)d_in[8];
  const float* dsk  = (const float*)d_in[9];
  const float* w_o  = (const float*)d_in[10];
  const float* nw   = (const float*)d_in[11];
  float* out = (float*)d_out;

  char* ws = (char*)d_ws;
  size_t off = 0;
  auto alloc = [&](size_t bytes) -> char* {
    char* p = ws + off; off = (off + bytes + 255) & ~(size_t)255; return p;
  };
  const size_t BLE = (size_t)ML_ * E_;
  const size_t BLD = (size_t)ML_ * DI_;
  float*          h_f  = (float*)alloc(BLE * 4);
  unsigned short* h_b  = (unsigned short*)alloc(BLE * 2);
  float*          xz_f = (float*)alloc(BLD * 2 * 4);
  float*          xc_f = (float*)alloc(BLD * 4);
  unsigned short* xc_b = (unsigned short*)alloc(BLD * 2);
  float*          xd_f = (float*)alloc((size_t)ML_ * 96 * 4);
  unsigned short* xd_b = (unsigned short*)alloc((size_t)ML_ * 96 * 2);
  float*          dl_f = (float*)alloc(BLD * 4);
  unsigned short* y_b  = (unsigned short*)alloc(BLD * 2);
  float*          o_f  = (float*)alloc(BLE * 4);
  unsigned short* w_in_b = (unsigned short*)alloc((size_t)NL_ * 2 * DI_ * E_ * 2);
  unsigned short* w_x_b  = (unsigned short*)alloc((size_t)NL_ * 96 * DI_ * 2);
  unsigned short* w_dt_b = (unsigned short*)alloc((size_t)NL_ * DI_ * DTR_ * 2);
  unsigned short* w_o_b  = (unsigned short*)alloc((size_t)NL_ * E_ * DI_ * 2);

  auto cvt = [&](const float* s, unsigned short* d, size_t n) {
    cvt_bf16_k<<<dim3((unsigned)((n + 255) / 256)), dim3(256), 0, stream>>>(s, d, (int)n);
  };
  cvt(w_in, w_in_b, (size_t)NL_ * 2 * DI_ * E_);
  cvt(w_x,  w_x_b,  (size_t)NL_ * 96 * DI_);
  cvt(w_dt, w_dt_b, (size_t)NL_ * DI_ * DTR_);
  cvt(w_o,  w_o_b,  (size_t)NL_ * E_ * DI_);

  addpos_k<<<dim3((unsigned)((BLE + 255) / 256)), 256, 0, stream>>>(x, pos, h_f, h_b, (int)BLE);

  for (int layer = 0; layer < NL_; ++layer) {
    // xz = h @ in_proj_w^T                     (M=4096, N=4096, K=1024)
    gemm_bf16_nt<<<dim3(ML_ / 64, (2 * DI_) / 256), 256, 0, stream>>>(
        h_b, E_, w_in_b + (size_t)layer * 2 * DI_ * E_, E_,
        xz_f, nullptr, 2 * DI_, ML_, 2 * DI_, E_, nullptr, 0);
    // causal depthwise conv + SiLU
    conv_silu_k<<<dim3((unsigned)((BLD + 255) / 256)), 256, 0, stream>>>(
        xz_f, cw + (size_t)layer * DI_ * DC_, cb + (size_t)layer * DI_, xc_f, xc_b);
    // x_dbl = xc @ x_proj_w^T                  (M=4096, N=96, K=2048)
    gemm_bf16_nt<<<dim3(ML_ / 64, 1), 256, 0, stream>>>(
        xc_b, DI_, w_x_b + (size_t)layer * 96 * DI_, DI_,
        xd_f, xd_b, 96, ML_, 96, DI_, nullptr, 0);
    // delta = softplus(dt @ dt_proj_w^T + b)   (M=4096, N=2048, K=64)
    gemm_bf16_nt<<<dim3(ML_ / 64, DI_ / 256), 256, 0, stream>>>(
        xd_b, 96, w_dt_b + (size_t)layer * DI_ * DTR_, DTR_,
        dl_f, nullptr, DI_, ML_, DI_, DTR_, dtb + (size_t)layer * DI_, 1);
    // selective scan + D-skip + SiLU gate -> y (bf16)
    scan_k<<<dim3(DI_ / 256, B_), 256, 0, stream>>>(
        dl_f, xd_f, xc_f, xz_f,
        alog + (size_t)layer * DI_ * DS_, dsk + (size_t)layer * DI_, y_b);
    // out = y @ out_proj_w^T                   (M=4096, N=1024, K=2048)
    gemm_bf16_nt<<<dim3(ML_ / 64, E_ / 256), 256, 0, stream>>>(
        y_b, DI_, w_o_b + (size_t)layer * E_ * DI_, DI_,
        o_f, nullptr, E_, ML_, E_, DI_, nullptr, 0);
    // h = rmsnorm(out + h) * norm_w
    rmsnorm_k<<<dim3(ML_), 256, 0, stream>>>(
        o_f, h_f, h_b, nw + (size_t)layer * E_, (layer == NL_ - 1) ? out : nullptr);
  }
}
